// ResourcePredictor_46815143526634
// MI455X (gfx1250) — matrix operand order repaired and checked by reading, not compiled, not run on hardware
//
#include <hip/hip_runtime.h>
#include <hip/hip_bf16.h>

#define BATCH   32768
#define SEQLEN  200
#define PREDS   50
#define HID     32
#define INDIM   4
#define ROWS_PER_WAVE   16
#define WAVES_PER_BLOCK 8
#define ROWS_PER_BLOCK  (ROWS_PER_WAVE * WAVES_PER_BLOCK)

typedef _Float16 v16h __attribute__((ext_vector_type(16)));
typedef _Float16 v8h  __attribute__((ext_vector_type(8)));
typedef float    v8f  __attribute__((ext_vector_type(8)));
typedef float    v2f  __attribute__((ext_vector_type(2)));

#define LOG2E 1.4426950408889634f

// tanh: native gfx1250 v_tanh_f32 (confirmed in round-2 disasm)
__device__ __forceinline__ float ftanh_(float x) {
#if __has_builtin(__builtin_amdgcn_tanhf)
  return __builtin_amdgcn_tanhf(x);
#else
  float e = __builtin_amdgcn_exp2f(-2.0f * LOG2E * x);
  return __builtin_fmaf(2.0f, __builtin_amdgcn_rcpf(1.0f + e), -1.0f);
#endif
}
// sigmoid(x) = 0.5*tanh(x/2) + 0.5 : mul + v_tanh_f32 + fma (1 trans op)
__device__ __forceinline__ float fsig(float x) {
#if __has_builtin(__builtin_amdgcn_tanhf)
  return __builtin_fmaf(0.5f, __builtin_amdgcn_tanhf(0.5f * x), 0.5f);
#else
  float e = __builtin_amdgcn_exp2f(-x * LOG2E);
  return __builtin_amdgcn_rcpf(1.0f + e);
#endif
}
__device__ __forceinline__ v8f bcast8(float x) {
  v8f v;
#pragma unroll
  for (int i = 0; i < 8; ++i) v[i] = x;
  return v;
}

// One wave (32 lanes) owns 16 batch rows.
//  A (h, f16 16x32):  lane l holds row M=(l&15); halves 0..7 = K {0..7 | 8..15},
//                     halves 8..15 = K {16..23 | 24..31} (low/high half-wave).
//  A (x, f32 16x4):   lane l holds row M=(l&15), 2 VGPRs = K {0,1 | 2,3}.
//  B (f16 32x16):     lane l holds col N=(l&15); halves 0..7 = K {0..7 | 8..15}, 8..15 = K {16..23 | 24..31}.
//  B (f32 4x16):      lane l holds col N=(l&15); 2 VGPRs = K {0,1 | 2,3}.
//  C/D (f32 16x16):   reg r, lane l -> (M = r (+8 high half), N = l&15).
__global__ __launch_bounds__(256, 1) void lstm_fused_kernel(
    const float* __restrict__ hist, const float* __restrict__ Wih,
    const float* __restrict__ Whh,  const float* __restrict__ bih,
    const float* __restrict__ bhh,  const float* __restrict__ Wpr,
    const float* __restrict__ bpr,  float* __restrict__ out)
{
  __shared__ __align__(16) _Float16 hbuf[WAVES_PER_BLOCK][ROWS_PER_WAVE][HID];
  __shared__ __align__(16) float    pbuf[WAVES_PER_BLOCK][ROWS_PER_WAVE][INDIM];

  const int lane = threadIdx.x & 31;
  const int wv   = threadIdx.x >> 5;
  const int col  = lane & 15;
  const int hiH  = lane >> 4;                       // 0 = lanes 0-15, 1 = lanes 16-31
  const int rowBase = blockIdx.x * ROWS_PER_BLOCK + wv * ROWS_PER_WAVE;
  const int m0   = hiH ? 8 : 0;
  const int aoff = hiH ? 8 : 0;

  // ---------- one-time weight staging into WMMA operand layouts ----------
  v16h Bhh[8];          // W_hh^T tiles, gate columns n*16+col, f16
  v2f  Bih[8];          // W_ih^T tiles, f32 (K = INDIM = 4)
  float bias[8];        // b_ih + b_hh per gate column
#pragma unroll
  for (int n = 0; n < 8; ++n) {
    const int j = n * 16 + col;                     // gate index 0..127 (i,f,g,o)
#pragma unroll
    for (int p = 0; p < 16; ++p) {
      const int K = (p < 8) ? 8 * hiH + p : 16 + 8 * hiH + (p - 8);
      Bhh[n][p] = (_Float16)Whh[j * HID + K];
    }
    const int k0 = hiH ? 2 : 0;
    Bih[n].x = Wih[j * INDIM + k0];
    Bih[n].y = Wih[j * INDIM + k0 + 1];
    bias[n] = bih[j] + bhh[j];
  }
  // W_pred as a zero-padded 32x16 f16 B tile (cols 0..3 valid)
  v16h Bp;
  float biasP = 0.0f;
#pragma unroll
  for (int p = 0; p < 16; ++p) {
    const int K = (p < 8) ? 8 * hiH + p : 16 + 8 * hiH + (p - 8);
    Bp[p] = (col < INDIM) ? (_Float16)Wpr[col * HID + K] : (_Float16)0.0f;
  }
  if (col < INDIM) biasP = bpr[col];

  // ---------- state ----------
  v16h Ah;                                          // h in A layout (f16)
#pragma unroll
  for (int p = 0; p < 16; ++p) Ah[p] = (_Float16)0.0f;
  v8f cst[2];                                       // c in C layout (two hidden halves)
#pragma unroll
  for (int q = 0; q < 2; ++q) cst[q] = bcast8(0.0f);

  const float* xrow =
      hist + (size_t)(rowBase + col) * (SEQLEN * INDIM) + (hiH ? 2 : 0);

  // ================= history encode: 200 sequential steps =================
  for (int t = 0; t < SEQLEN; ++t) {
    v2f Ax = *(const v2f*)(xrow + t * INDIM);       // x_t in A layout (f32)
    if (t + 1 < SEQLEN)
      __builtin_prefetch(xrow + (t + 1) * INDIM, 0, 3);

#pragma unroll
    for (int q = 0; q < 2; ++q) {                   // hidden halves 0-15 / 16-31
      v8f gi = __builtin_amdgcn_wmma_f32_16x16x32_f16(false, Ah, false, Bhh[q + 0], (short)0, bcast8(bias[q + 0]), false, false);
      v8f gf = __builtin_amdgcn_wmma_f32_16x16x32_f16(false, Ah, false, Bhh[q + 2], (short)0, bcast8(bias[q + 2]), false, false);
      v8f gg = __builtin_amdgcn_wmma_f32_16x16x32_f16(false, Ah, false, Bhh[q + 4], (short)0, bcast8(bias[q + 4]), false, false);
      v8f go = __builtin_amdgcn_wmma_f32_16x16x32_f16(false, Ah, false, Bhh[q + 6], (short)0, bcast8(bias[q + 6]), false, false);
      gi = __builtin_amdgcn_wmma_f32_16x16x4_f32(false, Ax, false, Bih[q + 0], (short)0, gi, false, false);
      gf = __builtin_amdgcn_wmma_f32_16x16x4_f32(false, Ax, false, Bih[q + 2], (short)0, gf, false, false);
      gg = __builtin_amdgcn_wmma_f32_16x16x4_f32(false, Ax, false, Bih[q + 4], (short)0, gg, false, false);
      go = __builtin_amdgcn_wmma_f32_16x16x4_f32(false, Ax, false, Bih[q + 6], (short)0, go, false, false);
#pragma unroll
      for (int r = 0; r < 8; ++r) {
        float cn = __builtin_fmaf(fsig(gf[r]), cst[q][r], fsig(gi[r]) * ftanh_(gg[r]));
        cst[q][r] = cn;
        float hn = fsig(go[r]) * ftanh_(cn);
        hbuf[wv][m0 + r][q * 16 + col] = (_Float16)hn;   // C layout -> LDS
      }
    }
    // same-wave DS ops are in-order; wait + clobber stops compile-time reorder
    asm volatile("s_wait_dscnt 0x0" ::: "memory");
    const _Float16* rp = &hbuf[wv][col][0];          // re-read h in A layout
    v8h c0 = *(const v8h*)(rp + aoff);
    v8h c1 = *(const v8h*)(rp + 16 + aoff);
#pragma unroll
    for (int p = 0; p < 8; ++p) { Ah[p] = c0[p]; Ah[8 + p] = c1[p]; }
  }

  // ================= autoregressive decode: 50 steps (c reset to 0) =================
  for (int s = 0; s < PREDS; ++s) {
    // pred = h @ W_pred^T + b_pred   (cols 0..3 of a 16-wide tile)
    v8f pr = __builtin_amdgcn_wmma_f32_16x16x32_f16(false, Ah, false, Bp, (short)0, bcast8(biasP), false, false);
    if (col < INDIM) {
#pragma unroll
      for (int r = 0; r < 8; ++r) {
        const int m = m0 + r;
        pbuf[wv][m][col] = pr[r];
        out[((size_t)(rowBase + m) * PREDS + s) * INDIM + col] = pr[r];
      }
    }
    asm volatile("s_wait_dscnt 0x0" ::: "memory");
    v2f Ax = *(const v2f*)(&pbuf[wv][col][0] + (hiH ? 2 : 0));  // pred -> A layout

#pragma unroll
    for (int q = 0; q < 2; ++q) {
      // c==0 each decode step, so sigmoid(f)*c == 0: skip the f-gate tiles.
      v8f gi = __builtin_amdgcn_wmma_f32_16x16x32_f16(false, Ah, false, Bhh[q + 0], (short)0, bcast8(bias[q + 0]), false, false);
      v8f gg = __builtin_amdgcn_wmma_f32_16x16x32_f16(false, Ah, false, Bhh[q + 4], (short)0, bcast8(bias[q + 4]), false, false);
      v8f go = __builtin_amdgcn_wmma_f32_16x16x32_f16(false, Ah, false, Bhh[q + 6], (short)0, bcast8(bias[q + 6]), false, false);
      gi = __builtin_amdgcn_wmma_f32_16x16x4_f32(false, Ax, false, Bih[q + 0], (short)0, gi, false, false);
      gg = __builtin_amdgcn_wmma_f32_16x16x4_f32(false, Ax, false, Bih[q + 4], (short)0, gg, false, false);
      go = __builtin_amdgcn_wmma_f32_16x16x4_f32(false, Ax, false, Bih[q + 6], (short)0, go, false, false);
#pragma unroll
      for (int r = 0; r < 8; ++r) {
        float cn = fsig(gi[r]) * ftanh_(gg[r]);
        float hn = fsig(go[r]) * ftanh_(cn);
        hbuf[wv][m0 + r][q * 16 + col] = (_Float16)hn;
      }
    }
    asm volatile("s_wait_dscnt 0x0" ::: "memory");
    const _Float16* rp = &hbuf[wv][col][0];
    v8h c0 = *(const v8h*)(rp + aoff);
    v8h c1 = *(const v8h*)(rp + 16 + aoff);
#pragma unroll
    for (int p = 0; p < 8; ++p) { Ah[p] = c0[p]; Ah[8 + p] = c1[p]; }
  }
}

extern "C" void kernel_launch(void* const* d_in, const int* in_sizes, int n_in,
                              void* d_out, int out_size, void* d_ws, size_t ws_size,
                              hipStream_t stream) {
  (void)in_sizes; (void)n_in; (void)out_size; (void)d_ws; (void)ws_size;
  const float* hist = (const float*)d_in[0];
  const float* Wih  = (const float*)d_in[1];
  const float* Whh  = (const float*)d_in[2];
  const float* bihp = (const float*)d_in[3];
  const float* bhhp = (const float*)d_in[4];
  const float* Wpr  = (const float*)d_in[5];
  const float* bpr  = (const float*)d_in[6];
  float* out = (float*)d_out;

  dim3 grid(BATCH / ROWS_PER_BLOCK);   // 256 blocks
  dim3 block(256);                     // 8 waves/block, 16 rows/wave
  hipLaunchKernelGGL(lstm_fused_kernel, grid, block, 0, stream,
                     hist, Wih, Whh, bihp, bhhp, Wpr, bpr, out);
}
